// MultiHeadDispatch_83580063580458
// MI455X (gfx1250) — matrix operand order repaired and checked by reading, not compiled, not run on hardware
//
#include <hip/hip_runtime.h>

// Problem dims (fixed by the reference)
#define B_    2
#define S_    2048
#define D_    1024
#define H_    16
#define DH_   64
#define MTOT  (B_ * S_)   // 4096

typedef __attribute__((ext_vector_type(16))) __bf16 v16bf;
typedef __attribute__((ext_vector_type(8)))  float  v8f;

union FragBF { v16bf v; unsigned u[8]; };

__device__ __forceinline__ v8f vzero() {
  v8f z = {0.f, 0.f, 0.f, 0.f, 0.f, 0.f, 0.f, 0.f};
  return z;
}

// fp32 -> bf16 bits, round-to-nearest-even
__device__ __forceinline__ unsigned short f2bf(float x) {
  unsigned u = __builtin_bit_cast(unsigned, x);
  u += 0x7FFFu + ((u >> 16) & 1u);
  return (unsigned short)(u >> 16);
}

// ---------------------------------------------------------------------------
// CDNA5 async global->LDS staging (ASYNCcnt path), guarded so a missing
// builtin falls back to plain 16B copies. Arg types per the compiler
// diagnostic: (int4 AS1*, int4 AS3*, imm offset, imm cpol).
// ---------------------------------------------------------------------------
#if defined(__AMDGCN__) && \
    __has_builtin(__builtin_amdgcn_global_load_async_to_lds_b128) && \
    __has_builtin(__builtin_amdgcn_s_wait_asynccnt)
#define USE_ASYNC 1
typedef __attribute__((ext_vector_type(4))) int v4i_t;
typedef __attribute__((address_space(1))) v4i_t* gas_v4i_p;
typedef __attribute__((address_space(3))) v4i_t* lds_v4i_p;
#else
#define USE_ASYNC 0
#endif

__device__ __forceinline__ void copy_b128(const unsigned short* g, unsigned short* l) {
#if USE_ASYNC
  __builtin_amdgcn_global_load_async_to_lds_b128((gas_v4i_p)g, (lds_v4i_p)l, 0, 0);
#else
  *(uint4*)l = *(const uint4*)g;
#endif
}

// Wait for our async tile DMAs, then block barrier.
__device__ __forceinline__ void wait_stage() {
#if USE_ASYNC
  __builtin_amdgcn_s_wait_asynccnt(0);
#endif
  __syncthreads();
}

// A-operand fragment (16x32 bf16, MxK): lane holds row M = lane%16.
// Per ISA 7.12.2: VGPR j (j<4): K = 8*half + 2j,2j+1 ; VGPR j (j>=4): K = 16 + 8*half + ...
__device__ __forceinline__ void frag_a(const unsigned short* row, int half, FragBF& f) {
#pragma unroll
  for (int j = 0; j < 4; ++j)
    f.u[j] = *reinterpret_cast<const unsigned*>(row + 8 * half + 2 * j);
#pragma unroll
  for (int j = 0; j < 4; ++j)
    f.u[4 + j] = *reinterpret_cast<const unsigned*>(row + 16 + 8 * half + 2 * j);
}

// B-operand fragment (32x16 bf16, KxN): lane holds column N = lane%16,
// same K order as frag_a: VGPR j (j<4): K = 8*half + 2j,2j+1 ; VGPR j (j>=4): K = 16 + 8*half + ...
__device__ __forceinline__ void frag_b(const unsigned short* row, int half, FragBF& f) {
#pragma unroll
  for (int j = 0; j < 4; ++j)
    f.u[j] = *reinterpret_cast<const unsigned*>(row + 8 * half + 2 * j);
#pragma unroll
  for (int j = 0; j < 4; ++j)
    f.u[4 + j] = *reinterpret_cast<const unsigned*>(row + 16 + 8 * half + 2 * j);
}

__device__ __forceinline__ v8f wmma_bf16(const FragBF& a, const FragBF& b, v8f c) {
  return __builtin_amdgcn_wmma_f32_16x16x32_bf16(
      false, a.v, false, b.v, (short)0, c, false, false);
}

// ---------------------------------------------------------------------------
// Kernel 0: one-shot fp32 -> bf16 conversion (X and the 4 weight matrices).
// ---------------------------------------------------------------------------
__global__ __launch_bounds__(256)
void conv_bf16_kernel(const float* __restrict__ src,
                      unsigned short* __restrict__ dst, int n) {
  int i = (blockIdx.x * blockDim.x + threadIdx.x) * 4;
  const int stride = gridDim.x * blockDim.x * 4;
  for (; i < n; i += stride) {
    float4 v = *(const float4*)(src + i);
    ushort4 o;
    o.x = f2bf(v.x); o.y = f2bf(v.y); o.z = f2bf(v.z); o.w = f2bf(v.w);
    *(ushort4*)(dst + i) = o;
  }
}

// ---------------------------------------------------------------------------
// Kernel 1: fused QKV projection on pre-converted bf16 operands.
// C[m,n] = X[m,:] . W[n,:] + bias[n]. grid.z selects Q/K/V.
// Q output: [B,H,S,DH] with 1/sqrt(DH) folded in. K: [B,H,S,DH]. V: [B,H,DH,S]
// (pre-transposed so attention can stage V^T tiles as straight copies).
// ---------------------------------------------------------------------------
__global__ __launch_bounds__(256)
void qkv_proj_kernel(const unsigned short* __restrict__ Xb,
                     const unsigned short* __restrict__ Wqb, const float* __restrict__ bq,
                     const unsigned short* __restrict__ Wkb, const float* __restrict__ bk,
                     const unsigned short* __restrict__ Wvb, const float* __restrict__ bv,
                     unsigned short* __restrict__ qws,
                     unsigned short* __restrict__ kws,
                     unsigned short* __restrict__ vws) {
  __shared__ unsigned short sA[2][128][32];  // 16 KB (double buffered)
  __shared__ unsigned short sB[2][128][32];  // 16 KB

  const int z = blockIdx.z;
  const unsigned short* W = (z == 0) ? Wqb : (z == 1 ? Wkb : Wvb);
  const float* bias = (z == 0) ? bq : (z == 1 ? bk : bv);
  unsigned short* out = (z == 0) ? qws : (z == 1 ? kws : vws);
  const float osc = (z == 0) ? 0.125f : 1.0f;

  const int m0 = blockIdx.y * 128;
  const int n0 = blockIdx.x * 128;
  const int t = threadIdx.x;
  const int w = t >> 5;
  const int lane = t & 31;
  const int half = lane >> 4;
  const int l16 = lane & 15;
  const int wm = w >> 2;  // 0..1 -> 64 rows
  const int wn = w & 3;   // 0..3 -> 32 cols

  // stage one 128x32 A tile + 128x32 B tile (4 b128 transfers per thread)
  auto stage = [&](int buf, int k0) {
#pragma unroll
    for (int i = 0; i < 2; ++i) {
      const int idx = t + i * 256;          // 0..511
      const int m = idx >> 2;               // 0..127
      const int kk = (idx & 3) * 8;         // 0,8,16,24
      copy_b128(Xb + (size_t)(m0 + m) * D_ + k0 + kk, &sA[buf][m][kk]);
      copy_b128(W  + (size_t)(n0 + m) * D_ + k0 + kk, &sB[buf][m][kk]);
    }
  };

  v8f acc[4][2];
#pragma unroll
  for (int i = 0; i < 4; ++i)
#pragma unroll
    for (int j = 0; j < 2; ++j) acc[i][j] = vzero();

  const int NK = D_ / 32;
  stage(0, 0);
  for (int ki = 0; ki < NK; ++ki) {
    wait_stage();                      // tile `cur` resident, prior reads done
    const int cur = ki & 1;
    if (ki + 1 < NK) stage(cur ^ 1, (ki + 1) * 32);  // overlap DMA with WMMA

    FragBF bfr[2];
#pragma unroll
    for (int nt = 0; nt < 2; ++nt)
      frag_b(&sB[cur][wn * 32 + nt * 16 + l16][0], half, bfr[nt]);
#pragma unroll
    for (int mt = 0; mt < 4; ++mt) {
      FragBF afr;
      frag_a(&sA[cur][wm * 64 + mt * 16 + l16][0], half, afr);
#pragma unroll
      for (int nt = 0; nt < 2; ++nt)
        acc[mt][nt] = wmma_bf16(afr, bfr[nt], acc[mt][nt]);
    }
  }

  // Epilogue: bias, scale, scatter as bf16.
#pragma unroll
  for (int mt = 0; mt < 4; ++mt) {
#pragma unroll
    for (int nt = 0; nt < 2; ++nt) {
      const int n = n0 + wn * 32 + nt * 16 + l16;
      const float bn = bias[n];
      const int h = n >> 6, dh = n & 63;
#pragma unroll
      for (int r = 0; r < 8; ++r) {
        const int m = m0 + wm * 64 + mt * 16 + r + 8 * half;
        const int b = m >> 11;      // m / S_
        const int s = m & (S_ - 1);
        const float val = (acc[mt][nt][r] + bn) * osc;
        const size_t bh = (size_t)(b * H_ + h);
        const size_t idx = (z == 2) ? (bh * DH_ + dh) * S_ + s    // V transposed
                                    : (bh * S_ + s) * DH_ + dh;   // Q/K
        out[idx] = f2bf(val);
      }
    }
  }
}

// ---------------------------------------------------------------------------
// Kernel 2: flash attention per (b,h). 128 q-rows per block, 8 waves x 16 rows.
// K tiles [key][dh] and V^T tiles [dh][key] staged async, double buffered.
// ---------------------------------------------------------------------------
__global__ __launch_bounds__(256)
void attn_kernel(const unsigned short* __restrict__ qws,
                 const unsigned short* __restrict__ kws,
                 const unsigned short* __restrict__ vws,   // [B,H,DH,S]
                 unsigned short* __restrict__ yws) {
  __shared__ unsigned short sQ[128][64];       // 16 KB
  __shared__ unsigned short sK[2][32][64];     //  8 KB
  __shared__ unsigned short sVT[2][64][32];    //  8 KB
  __shared__ unsigned short sP[8][16][32];     //  8 KB

  const int bh = blockIdx.y;  // b*H + h
  const int b = bh >> 4;
  const int h = bh & 15;
  const int q0 = blockIdx.x * 128;
  const int t = threadIdx.x;
  const int w = t >> 5;
  const int lane = t & 31;
  const int half = lane >> 4;
  const int l16 = lane & 15;

  const size_t base = (size_t)bh * S_ * DH_;
  const unsigned short* vt = vws + (size_t)bh * DH_ * S_;

  // stage Q (contiguous 16 KB span), 4 transfers per thread
#pragma unroll
  for (int i = 0; i < 4; ++i) {
    const int idx = t + i * 256;
    copy_b128(qws + base + (size_t)q0 * DH_ + idx * 8, &sQ[0][0] + idx * 8);
  }

  auto stageKV = [&](int buf, int kb) {
    // K tile: contiguous 4 KB span, 1 transfer/thread
    copy_b128(kws + base + (size_t)kb * DH_ + t * 8, &sK[buf][0][0] + t * 8);
    // V^T tile: 64 rows of 64B, 1 transfer/thread
    const int dh = t >> 2, kk = (t & 3) * 8;
    copy_b128(vt + (size_t)dh * S_ + kb + kk, &sVT[buf][dh][kk]);
  };

  stageKV(0, 0);
  wait_stage();

  FragBF aQ[2];
#pragma unroll
  for (int ks = 0; ks < 2; ++ks)
    frag_a(&sQ[w * 16 + l16][ks * 32], half, aQ[ks]);

  v8f yacc[4];
#pragma unroll
  for (int i = 0; i < 4; ++i) yacc[i] = vzero();
  float rmax[8], rsum[8];
#pragma unroll
  for (int r = 0; r < 8; ++r) { rmax[r] = -3.0e38f; rsum[r] = 0.0f; }

  const float LOG2E = 1.4426950408889634f;
  const int NKB = S_ / 32;

  for (int ki = 0; ki < NKB; ++ki) {
    if (ki) wait_stage();
    const int cur = ki & 1;
    if (ki + 1 < NKB) stageKV(cur ^ 1, (ki + 1) * 32);  // overlap DMA

    // scores: 16x32 per wave (softmax scale already folded into Q)
    v8f c[2];
#pragma unroll
    for (int nt = 0; nt < 2; ++nt) {
      c[nt] = vzero();
#pragma unroll
      for (int ks = 0; ks < 2; ++ks) {
        FragBF bK;
        frag_b(&sK[cur][nt * 16 + l16][ks * 32], half, bK);
        c[nt] = wmma_bf16(aQ[ks], bK, c[nt]);
      }
    }

    // online softmax (row r+8*half spread across 16 lanes -> xor 1/2/4/8)
    float al[8];
#pragma unroll
    for (int r = 0; r < 8; ++r) {
      float m = fmaxf(c[0][r], c[1][r]);
#pragma unroll
      for (int off = 1; off < 16; off <<= 1)
        m = fmaxf(m, __shfl_xor(m, off, 32));
      const float nm = fmaxf(rmax[r], m);
      al[r] = exp2f((rmax[r] - nm) * LOG2E);
      const float p0 = exp2f((c[0][r] - nm) * LOG2E);
      const float p1 = exp2f((c[1][r] - nm) * LOG2E);
      float rs = p0 + p1;
#pragma unroll
      for (int off = 1; off < 16; off <<= 1)
        rs += __shfl_xor(rs, off, 32);
      rsum[r] = rsum[r] * al[r] + rs;
      rmax[r] = nm;
      sP[w][r + 8 * half][l16]      = f2bf(p0);  // spill P to A-layout staging
      sP[w][r + 8 * half][16 + l16] = f2bf(p1);
    }
#pragma unroll
    for (int i = 0; i < 4; ++i)
#pragma unroll
      for (int r = 0; r < 8; ++r) yacc[i][r] *= al[r];

    __syncthreads();  // sP visible

    FragBF aP;
    frag_a(&sP[w][l16][0], half, aP);
#pragma unroll
    for (int nt = 0; nt < 4; ++nt) {
      FragBF bV;
      frag_b(&sVT[cur][nt * 16 + l16][0], half, bV);
      yacc[nt] = wmma_bf16(aP, bV, yacc[nt]);
    }
  }

  // normalize and store merged-head y as bf16 [B,S,D]
#pragma unroll
  for (int r = 0; r < 8; ++r) {
    const float inv = 1.0f / rsum[r];
    const int s = q0 + w * 16 + r + 8 * half;
#pragma unroll
    for (int nt = 0; nt < 4; ++nt) {
      const int dh = nt * 16 + l16;
      yws[((size_t)(b * S_ + s)) * D_ + h * DH_ + dh] = f2bf(yacc[nt][r] * inv);
    }
  }
}

// ---------------------------------------------------------------------------
// Kernel 3: output projection. C = Y(bf16) . Wo^T + bo, fp32 output.
// ---------------------------------------------------------------------------
__global__ __launch_bounds__(256)
void out_proj_kernel(const unsigned short* __restrict__ Y,
                     const unsigned short* __restrict__ Wob,
                     const float* __restrict__ bo,
                     float* __restrict__ out) {
  __shared__ unsigned short sA[2][128][32];
  __shared__ unsigned short sB[2][128][32];

  const int m0 = blockIdx.y * 128;
  const int n0 = blockIdx.x * 128;
  const int t = threadIdx.x;
  const int w = t >> 5;
  const int lane = t & 31;
  const int half = lane >> 4;
  const int l16 = lane & 15;
  const int wm = w >> 2;
  const int wn = w & 3;

  auto stage = [&](int buf, int k0) {
#pragma unroll
    for (int i = 0; i < 2; ++i) {
      const int idx = t + i * 256;
      const int m = idx >> 2;
      const int kk = (idx & 3) * 8;
      copy_b128(Y   + (size_t)(m0 + m) * D_ + k0 + kk, &sA[buf][m][kk]);
      copy_b128(Wob + (size_t)(n0 + m) * D_ + k0 + kk, &sB[buf][m][kk]);
    }
  };

  v8f acc[4][2];
#pragma unroll
  for (int i = 0; i < 4; ++i)
#pragma unroll
    for (int j = 0; j < 2; ++j) acc[i][j] = vzero();

  const int NK = D_ / 32;
  stage(0, 0);
  for (int ki = 0; ki < NK; ++ki) {
    wait_stage();
    const int cur = ki & 1;
    if (ki + 1 < NK) stage(cur ^ 1, (ki + 1) * 32);

    FragBF bfr[2];
#pragma unroll
    for (int nt = 0; nt < 2; ++nt)
      frag_b(&sB[cur][wn * 32 + nt * 16 + l16][0], half, bfr[nt]);
#pragma unroll
    for (int mt = 0; mt < 4; ++mt) {
      FragBF afr;
      frag_a(&sA[cur][wm * 64 + mt * 16 + l16][0], half, afr);
#pragma unroll
      for (int nt = 0; nt < 2; ++nt)
        acc[mt][nt] = wmma_bf16(afr, bfr[nt], acc[mt][nt]);
    }
  }

#pragma unroll
  for (int mt = 0; mt < 4; ++mt) {
#pragma unroll
    for (int nt = 0; nt < 2; ++nt) {
      const int n = n0 + wn * 32 + nt * 16 + l16;
      const float bn = bo[n];
#pragma unroll
      for (int r = 0; r < 8; ++r) {
        const int m = m0 + wm * 64 + mt * 16 + r + 8 * half;
        out[(size_t)m * D_ + n] = acc[mt][nt][r] + bn;
      }
    }
  }
}

// ---------------------------------------------------------------------------
extern "C" void kernel_launch(void* const* d_in, const int* in_sizes, int n_in,
                              void* d_out, int out_size, void* d_ws, size_t ws_size,
                              hipStream_t stream) {
  const float* X  = (const float*)d_in[0];
  const float* Wq = (const float*)d_in[1];
  const float* bq = (const float*)d_in[2];
  const float* Wk = (const float*)d_in[3];
  const float* bk = (const float*)d_in[4];
  const float* Wv = (const float*)d_in[5];
  const float* bv = (const float*)d_in[6];
  const float* Wo = (const float*)d_in[7];
  const float* bo = (const float*)d_in[8];
  float* out = (float*)d_out;

  const size_t nX = (size_t)MTOT * D_;      // 4,194,304
  const size_t nW = (size_t)D_ * D_;        // 1,048,576
  unsigned short* p = (unsigned short*)d_ws;
  unsigned short* Xb  = p; p += nX;
  unsigned short* Wqb = p; p += nW;
  unsigned short* Wkb = p; p += nW;
  unsigned short* Wvb = p; p += nW;
  unsigned short* Wob = p; p += nW;
  unsigned short* qws = p; p += nX;
  unsigned short* kws = p; p += nX;
  unsigned short* vws = p; p += nX;         // stored transposed [B,H,DH,S]
  unsigned short* yws = p;                  // total ~48 MB

  dim3 blk(256);
  conv_bf16_kernel<<<dim3(512), blk, 0, stream>>>(X,  Xb,  (int)nX);
  conv_bf16_kernel<<<dim3(512), blk, 0, stream>>>(Wq, Wqb, (int)nW);
  conv_bf16_kernel<<<dim3(512), blk, 0, stream>>>(Wk, Wkb, (int)nW);
  conv_bf16_kernel<<<dim3(512), blk, 0, stream>>>(Wv, Wvb, (int)nW);
  conv_bf16_kernel<<<dim3(512), blk, 0, stream>>>(Wo, Wob, (int)nW);

  qkv_proj_kernel<<<dim3(D_ / 128, MTOT / 128, 3), blk, 0, stream>>>(
      Xb, Wqb, bq, Wkb, bk, Wvb, bv, qws, kws, vws);
  attn_kernel<<<dim3(S_ / 128, B_ * H_), blk, 0, stream>>>(qws, kws, vws, yws);
  out_proj_kernel<<<dim3(D_ / 128, MTOT / 128), blk, 0, stream>>>(yws, Wob, bo, out);
}
